// EdgeProcessor_6133213298853
// MI455X (gfx1250) — matrix operand order repaired and checked by reading, not compiled, not run on hardware
//
#include <hip/hip_runtime.h>

typedef _Float16 v16h __attribute__((ext_vector_type(16)));
typedef _Float16 v8h  __attribute__((ext_vector_type(8)));
typedef float    v8f  __attribute__((ext_vector_type(8)));
typedef float    v4f  __attribute__((ext_vector_type(4)));

#define E_TOTAL 262144
#define D_DIM   128
#define H_DIM   512
#define IN_DIM  384
#define M_TILE  128
#define THREADS 256
#define LDA (IN_DIM + 8)   // 392 halves, padded (rotating bank groups per row)
#define LDH (H_DIM + 8)    // 520 halves
#define LDC (D_DIM + 4)    // 132 floats
#define KB1 (IN_DIM / 32)  // 12
#define KB2 (H_DIM / 32)   // 16
#define FRAG_H 512         // halves per fragment (32 lanes * 16 halves)

__device__ __forceinline__ v16h cat8(v8h lo, v8h hi) {
  return __builtin_shufflevector(lo, hi, 0,1,2,3,4,5,6,7,8,9,10,11,12,13,14,15);
}

__device__ __forceinline__ v8f wmma_f16(v16h a, v16h b, v8f c) {
  return __builtin_amdgcn_wmma_f32_16x16x32_f16(false, a, false, b, (short)0, c, false, false);
}

// ---------------------------------------------------------------------------
// Pack W1/W2 (fp32 row-major [K,N]) into f16 WMMA B-fragment layout:
// frag (kb,nb) at halves offset (nb*KB + kb)*512; per lane l, 16 contiguous
// halves j=0..15:  n = nb*16 + (l & 15),  k = kb*32 + ((j>>3)<<4) + ((l>>4)<<3) + (j&7)
// ---------------------------------------------------------------------------
__global__ void pack_weights(const float* __restrict__ W1,
                             const float* __restrict__ W2,
                             _Float16* __restrict__ w1p,
                             _Float16* __restrict__ w2p) {
  int tid = blockIdx.x * blockDim.x + threadIdx.x;
  if (tid < IN_DIM * H_DIM) {
    int j    = tid & 15;
    int lane = (tid >> 4) & 31;
    int frag = tid >> 9;
    int kb = frag % KB1, nb = frag / KB1;
    int n = nb * 16 + (lane & 15);
    int k = kb * 32 + ((j >> 3) << 4) + ((lane >> 4) << 3) + (j & 7);
    w1p[tid] = (_Float16)W1[k * H_DIM + n];
  } else {
    int t    = tid - IN_DIM * H_DIM;
    int j    = t & 15;
    int lane = (t >> 4) & 31;
    int frag = t >> 9;
    int kb = frag % KB2, nb = frag / KB2;
    int n = nb * 16 + (lane & 15);
    int k = kb * 32 + ((j >> 3) << 4) + ((lane >> 4) << 3) + (j & 7);
    w2p[t] = (_Float16)W2[k * D_DIM + n];
  }
}

// ---------------------------------------------------------------------------
// Fused edge MLP: concat -> GEMM1(384x512)+SiLU -> GEMM2(512x128)
//                 -> LayerNorm -> +edge_attr
// One block = 128 edges, 8 waves. Per wave: 32x64 register tile per pass
// (2 M-frags x 4 N-frags, f32 accum). B fragments read via ONE per-lane base
// pointer + constant strides so loads fold into immediate offsets.
// ---------------------------------------------------------------------------
__global__ void __launch_bounds__(THREADS, 1)
edge_mlp_kernel(const float* __restrict__ x_i,
                const float* __restrict__ x_j,
                const float* __restrict__ edge_attr,
                const float* __restrict__ b1,
                const float* __restrict__ b2,
                const float* __restrict__ gamma,
                const float* __restrict__ beta,
                const _Float16* __restrict__ w1p,
                const _Float16* __restrict__ w2p,
                float* __restrict__ out) {
  __shared__ __align__(16) _Float16 A1[M_TILE * LDA];  // 100,352 B (reused as f32 H2)
  __shared__ __align__(16) _Float16 H1[M_TILE * LDH];  // 133,120 B
  float* H2 = (float*)A1;                              // stride LDC floats

  const int tid     = threadIdx.x;
  const int lane    = tid & 31;
  const int wave    = tid >> 5;
  const int r0      = blockIdx.x * M_TILE;
  const int mlane   = lane & 15;
  const int halfSel = lane >> 4;          // 0|1
  const int baseKA  = halfSel * 8;        // A-fragment K sub-offset

  // ---- Phase 0: stage concat([x_i,x_j,edge_attr]) as f16 into LDS ----
  for (int idx = tid; idx < M_TILE * (IN_DIM / 8); idx += THREADS) {
    int row = idx / (IN_DIM / 8);
    int col = (idx % (IN_DIM / 8)) * 8;
    const float* src;
    if (col < D_DIM)          src = x_i       + (r0 + row) * D_DIM + col;
    else if (col < 2 * D_DIM) src = x_j       + (r0 + row) * D_DIM + (col - D_DIM);
    else                      src = edge_attr + (r0 + row) * D_DIM + (col - 2 * D_DIM);
    v4f v0 = *(const v4f*)src;
    v4f v1 = *(const v4f*)(src + 4);
    v8h h = { (_Float16)v0.x, (_Float16)v0.y, (_Float16)v0.z, (_Float16)v0.w,
              (_Float16)v1.x, (_Float16)v1.y, (_Float16)v1.z, (_Float16)v1.w };
    *(v8h*)&A1[row * LDA + col] = h;
  }
  __syncthreads();

  const int wm = wave & 3;   // 4 row-groups of 32 rows
  const int wn = wave >> 2;  // 2 col-halves
  const int rowA0 = wm * 32 + mlane;
  const int rowA1 = rowA0 + 16;

  // ---- Phase 1: GEMM1 + bias + SiLU -> H1 (f16) ----
  for (int c = 0; c < 4; ++c) {
    const int n0 = wn * 256 + c * 64;
    // single per-lane B base; fragment (nb0+f, kb) at +f*KB1*512 + kb*512 halves
    const _Float16* pb = w1p + (size_t)(n0 >> 4) * KB1 * FRAG_H + lane * 16;
    const _Float16* pa0 = &A1[rowA0 * LDA + baseKA];
    const _Float16* pa1 = &A1[rowA1 * LDA + baseKA];
    v8f acc[2][4] = {};
    #pragma unroll 2
    for (int kb = 0; kb < KB1; ++kb) {
      v16h a0 = cat8(*(const v8h*)pa0, *(const v8h*)(pa0 + 16));
      v16h a1 = cat8(*(const v8h*)pa1, *(const v8h*)(pa1 + 16));
      #pragma unroll
      for (int f = 0; f < 4; ++f) {
        v16h b = *(const v16h*)(pb + f * (KB1 * FRAG_H));  // immediate offsets
        acc[0][f] = wmma_f16(a0, b, acc[0][f]);
        acc[1][f] = wmma_f16(a1, b, acc[1][f]);
      }
      pa0 += 32; pa1 += 32; pb += FRAG_H;
    }
    #pragma unroll
    for (int mr = 0; mr < 2; ++mr) {
      int mbase = wm * 32 + mr * 16 + halfSel * 8;
      #pragma unroll
      for (int f = 0; f < 4; ++f) {
        int n = n0 + f * 16 + mlane;
        float bias = b1[n];
        #pragma unroll
        for (int r = 0; r < 8; ++r) {
          float v = acc[mr][f][r] + bias;
          float s = v * __builtin_amdgcn_rcpf(1.0f + __expf(-v));  // SiLU
          H1[(mbase + r) * LDH + n] = (_Float16)s;
        }
      }
    }
  }
  __syncthreads();

  // ---- Phase 2: GEMM2 + bias -> H2 (f32, aliases A1 which is dead) ----
  {
    const int n0 = wn * 64;
    const _Float16* pb = w2p + (size_t)(n0 >> 4) * KB2 * FRAG_H + lane * 16;
    const _Float16* pa0 = &H1[rowA0 * LDH + baseKA];
    const _Float16* pa1 = &H1[rowA1 * LDH + baseKA];
    v8f acc[2][4] = {};
    #pragma unroll 2
    for (int kb = 0; kb < KB2; ++kb) {
      v16h a0 = cat8(*(const v8h*)pa0, *(const v8h*)(pa0 + 16));
      v16h a1 = cat8(*(const v8h*)pa1, *(const v8h*)(pa1 + 16));
      #pragma unroll
      for (int f = 0; f < 4; ++f) {
        v16h b = *(const v16h*)(pb + f * (KB2 * FRAG_H));  // immediate offsets
        acc[0][f] = wmma_f16(a0, b, acc[0][f]);
        acc[1][f] = wmma_f16(a1, b, acc[1][f]);
      }
      pa0 += 32; pa1 += 32; pb += FRAG_H;
    }
    #pragma unroll
    for (int mr = 0; mr < 2; ++mr) {
      int mbase = wm * 32 + mr * 16 + halfSel * 8;
      #pragma unroll
      for (int f = 0; f < 4; ++f) {
        int n = n0 + f * 16 + mlane;
        float bias = b2[n];
        #pragma unroll
        for (int r = 0; r < 8; ++r) {
          H2[(mbase + r) * LDC + n] = acc[mr][f][r] + bias;
        }
      }
    }
  }
  __syncthreads();

  // ---- Phase 3: LayerNorm over D=128 + residual, 16 rows per wave ----
  const int c0 = lane * 4;
  const v4f g4  = *(const v4f*)&gamma[c0];
  const v4f be4 = *(const v4f*)&beta[c0];
  for (int rr = 0; rr < 16; ++rr) {
    int m = wave * 16 + rr;
    v4f x = *(const v4f*)&H2[m * LDC + c0];
    float s1 = x.x + x.y + x.z + x.w;
    float s2 = x.x * x.x + x.y * x.y + x.z * x.z + x.w * x.w;
    #pragma unroll
    for (int off = 16; off >= 1; off >>= 1) {
      s1 += __shfl_xor(s1, off, 32);
      s2 += __shfl_xor(s2, off, 32);
    }
    float mu  = s1 * (1.0f / 128.0f);
    float var = s2 * (1.0f / 128.0f) - mu * mu;
    float rs  = __builtin_amdgcn_rsqf(var + 1e-5f);
    v4f e = *(const v4f*)&edge_attr[(size_t)(r0 + m) * D_DIM + c0];
    v4f o;
    o.x = (x.x - mu) * rs * g4.x + be4.x + e.x;
    o.y = (x.y - mu) * rs * g4.y + be4.y + e.y;
    o.z = (x.z - mu) * rs * g4.z + be4.z + e.z;
    o.w = (x.w - mu) * rs * g4.w + be4.w + e.w;
    *(v4f*)&out[(size_t)(r0 + m) * D_DIM + c0] = o;
  }
}

extern "C" void kernel_launch(void* const* d_in, const int* in_sizes, int n_in,
                              void* d_out, int out_size, void* d_ws, size_t ws_size,
                              hipStream_t stream) {
  const float* x_i       = (const float*)d_in[0];
  const float* x_j       = (const float*)d_in[1];
  const float* edge_attr = (const float*)d_in[2];
  const float* W1        = (const float*)d_in[3];
  const float* b1        = (const float*)d_in[4];
  const float* W2        = (const float*)d_in[5];
  const float* b2        = (const float*)d_in[6];
  const float* gamma     = (const float*)d_in[7];
  const float* beta      = (const float*)d_in[8];

  _Float16* w1p = (_Float16*)d_ws;                                       // 393,216 B
  _Float16* w2p = (_Float16*)((char*)d_ws + (size_t)IN_DIM * H_DIM * 2); // +131,072 B

  const int packElems = IN_DIM * H_DIM + H_DIM * D_DIM;                  // 262,144
  pack_weights<<<packElems / 256, 256, 0, stream>>>(W1, W2, w1p, w2p);

  edge_mlp_kernel<<<E_TOTAL / M_TILE, THREADS, 0, stream>>>(
      x_i, x_j, edge_attr, b1, b2, gamma, beta, w1p, w2p, (float*)d_out);
}
